// MultiheadGeneticAttention_10969346474142
// MI455X (gfx1250) — matrix operand order repaired and checked by reading, not compiled, not run on hardware
//
#include <hip/hip_runtime.h>
#include <hip/hip_bf16.h>
#include <cstdint>

// ---------------------------------------------------------------------------
// MI455X / gfx1250. wave32, WMMA f32_16x16x32_f16, async global->LDS copies.
// B=2 S=2048 E=1024 H=16 HKV=4 D=64
// ---------------------------------------------------------------------------

typedef __attribute__((ext_vector_type(16))) _Float16 v16h;
typedef __attribute__((ext_vector_type(8)))  float    v8f;
typedef __attribute__((ext_vector_type(8)))  uint32_t u32x8;
typedef __attribute__((ext_vector_type(4)))  uint32_t u32x4;
typedef __attribute__((ext_vector_type(2)))  __fp16   fp16x2;

__device__ __forceinline__ uint32_t pack2(float a, float b) {
    fp16x2 r = __builtin_amdgcn_cvt_pkrtz(a, b);
    return __builtin_bit_cast(uint32_t, r);
}
__device__ __forceinline__ uint16_t f16bits(float a) {
    return __builtin_bit_cast(uint16_t, (_Float16)a);
}

__device__ __forceinline__ v8f wmma_f16(v16h a, v16h b, v8f c) {
    return __builtin_amdgcn_wmma_f32_16x16x32_f16(
        false, a, false, b, (short)0, c, false, false);
}

// Async global->LDS 16-byte copy per lane (gfx1250, ASYNCcnt-tracked).
__device__ __forceinline__ void async_b128(uint32_t lds_byte_off, uint64_t gaddr) {
    asm volatile("global_load_async_to_lds_b128 %0, %1, off"
                 :: "v"(lds_byte_off), "v"(gaddr) : "memory");
}
__device__ __forceinline__ void wait_async0() {
    asm volatile("s_wait_asynccnt 0" ::: "memory");
}
__device__ __forceinline__ void wait_async2() {
    asm volatile("s_wait_asynccnt 2" ::: "memory");
}
__device__ __forceinline__ void wait_async4() {
    asm volatile("s_wait_asynccnt 4" ::: "memory");
}

// A-fragment (16x32 f16): lane l holds row m=l&15; two 4-dword runs.
__device__ __forceinline__ v16h load_afrag(const uint32_t* rowp, int lane) {
    const int o = (lane >> 4) << 2;
    u32x4 lo = *(const u32x4*)(rowp + o);
    u32x4 hi = *(const u32x4*)(rowp + 8 + o);
    u32x8 d;
    d[0] = lo[0]; d[1] = lo[1]; d[2] = lo[2]; d[3] = lo[3];
    d[4] = hi[0]; d[5] = hi[1]; d[6] = hi[2]; d[7] = hi[3];
    return __builtin_bit_cast(v16h, d);
}
// B-fragment (32x16 f16): lane l: n=l&15; two 4-dword runs (same k order as A).
__device__ __forceinline__ v16h load_bfrag(const uint32_t* colp, int lane) {
    const int o = (lane >> 4) << 2;
    u32x4 lo = *(const u32x4*)(colp + o);
    u32x4 hi = *(const u32x4*)(colp + 8 + o);
    u32x8 d;
    d[0] = lo[0]; d[1] = lo[1]; d[2] = lo[2]; d[3] = lo[3];
    d[4] = hi[0]; d[5] = hi[1]; d[6] = hi[2]; d[7] = hi[3];
    return __builtin_bit_cast(v16h, d);
}

// ---------------------------------------------------------------------------
// Prep: fp32 -> f16 flat copy (2 elements / thread).
// ---------------------------------------------------------------------------
__global__ __launch_bounds__(256) void f32_to_f16_kernel(
    const float* __restrict__ in, uint32_t* __restrict__ out, int n2)
{
    int i = blockIdx.x * 256 + threadIdx.x;
    if (i < n2) {
        const float2 f = *(const float2*)(in + (size_t)i * 2);
        out[i] = pack2(f.x, f.y);
    }
}

// Prep: W (K,N) fp32 -> Wt (N,K) f16 (transposed, so GEMM B-tiles are copies).
__global__ __launch_bounds__(256) void transpose_f16_kernel(
    const float* __restrict__ W, uint16_t* __restrict__ Wt, int K, int N)
{
    int id = blockIdx.x * 256 + threadIdx.x;
    if (id < K * N) {
        int k = id / N, n = id % N;
        Wt[(size_t)n * K + k] = f16bits(W[id]);
    }
}

// Prep: vraw (B,S,HKV,64) fp32 -> vt (B,HKV,64,S) f16 (d-major for PV B-tiles).
__global__ __launch_bounds__(256) void vpack_t_kernel(
    const float* __restrict__ vraw, uint16_t* __restrict__ vt, int S, int HKVc)
{
    int id = blockIdx.x * 256 + threadIdx.x;
    int d = id & 63;
    int rest = id >> 6;
    int kv = rest % HKVc; rest /= HKVc;
    int s = rest % S;
    int b = rest / S;
    vt[((size_t)(b * HKVc + kv) * 64 + d) * S + s] = f16bits(vraw[id]);
}

// ---------------------------------------------------------------------------
// GEMM: C[M,N] fp32 = Ah[M,K] f16 @ Wt[N,K] f16 (+bias). Block tile 128x128,
// 8 waves 4x2, each wave 32x64 (8 C-frags). Double-buffered async LDS tiles.
// ---------------------------------------------------------------------------
__global__ __launch_bounds__(256) void gemm_wmma_kernel(
    const uint16_t* __restrict__ Ah, const uint16_t* __restrict__ Wt,
    const float* __restrict__ bias, float* __restrict__ C,
    int M, int N, int K)
{
    __shared__ __align__(16) uint32_t Abuf[2][128 * 16]; // [m][16 dw] per buf
    __shared__ __align__(16) uint32_t Bbuf[2][128 * 16]; // [n][16 dw] per buf

    const int t    = threadIdx.x;
    const int lane = t & 31;
    const int wave = t >> 5;
    const int wm   = wave >> 1;  // 0..3 : 32-row group
    const int wn   = wave & 1;   // 0..1 : 64-col group
    const int m0   = blockIdx.x * 128;
    const int n0   = blockIdx.y * 128;

    const uint32_t aBase = (uint32_t)(uintptr_t)&Abuf[0][0];
    const uint32_t bBase = (uint32_t)(uintptr_t)&Bbuf[0][0];
    const int seg = t & 3;
    const uint64_t gA0 = (uint64_t)(uintptr_t)Ah + ((size_t)(m0 + (t >> 2))      * K) * 2 + seg * 16;
    const uint64_t gA1 = (uint64_t)(uintptr_t)Ah + ((size_t)(m0 + (t >> 2) + 64) * K) * 2 + seg * 16;
    const uint64_t gB0 = (uint64_t)(uintptr_t)Wt + ((size_t)(n0 + (t >> 2))      * K) * 2 + seg * 16;
    const uint64_t gB1 = (uint64_t)(uintptr_t)Wt + ((size_t)(n0 + (t >> 2) + 64) * K) * 2 + seg * 16;
    const uint32_t ldsA0 = aBase + (uint32_t)t * 16;
    const uint32_t ldsA1 = aBase + (uint32_t)(t + 256) * 16;
    const uint32_t ldsB0 = bBase + (uint32_t)t * 16;
    const uint32_t ldsB1 = bBase + (uint32_t)(t + 256) * 16;

    v8f acc[2][4];
    #pragma unroll
    for (int i = 0; i < 2; ++i)
        #pragma unroll
        for (int j = 0; j < 4; ++j)
            acc[i][j] = (v8f){0.f,0.f,0.f,0.f,0.f,0.f,0.f,0.f};

    auto issue = [&](int buf, int k0) {
        const uint32_t bo = (uint32_t)buf * (128 * 16 * 4);
        const uint64_t go = (uint64_t)k0 * 2;
        async_b128(ldsA0 + bo, gA0 + go);
        async_b128(ldsA1 + bo, gA1 + go);
        async_b128(ldsB0 + bo, gB0 + go);
        async_b128(ldsB1 + bo, gB1 + go);
    };
    auto compute = [&](int cur) {
        v16h a0 = load_afrag(&Abuf[cur][(wm * 32      + (lane & 15)) * 16], lane);
        v16h a1 = load_afrag(&Abuf[cur][(wm * 32 + 16 + (lane & 15)) * 16], lane);
        #pragma unroll
        for (int ni = 0; ni < 4; ++ni) {
            v16h b = load_bfrag(&Bbuf[cur][(wn * 64 + ni * 16 + (lane & 15)) * 16], lane);
            acc[0][ni] = wmma_f16(a0, b, acc[0][ni]);
            acc[1][ni] = wmma_f16(a1, b, acc[1][ni]);
        }
    };

    // Pipeline: issue tile 0; steady-state loop issues next tile, waits for
    // the current one (<=4 leaves only the just-issued batch outstanding).
    issue(0, 0);
    for (int k0 = 0; k0 < K - 32; k0 += 32) {
        const int cur = (k0 >> 5) & 1;
        issue(cur ^ 1, k0 + 32);
        wait_async4();
        __syncthreads();
        compute(cur);
        __syncthreads();
    }
    {
        const int cur = ((K - 32) >> 5) & 1;
        wait_async0();
        __syncthreads();
        compute(cur);
    }

    #pragma unroll
    for (int mi = 0; mi < 2; ++mi) {
        #pragma unroll
        for (int ni = 0; ni < 4; ++ni) {
            int n = n0 + wn * 64 + ni * 16 + (lane & 15);
            float bv = bias[n];
            #pragma unroll
            for (int r = 0; r < 8; ++r) {
                int m = m0 + wm * 32 + mi * 16 + r + ((lane >> 4) << 3);
                C[(size_t)m * N + n] = acc[mi][ni][r] + bv;
            }
        }
    }
}

// ---------------------------------------------------------------------------
// RMSNorm + relayout (B,S,Hn,64) fp32 -> (B,Hn,S,64) f16. One wave per row.
// ---------------------------------------------------------------------------
__global__ __launch_bounds__(256) void rmsnorm_pack_kernel(
    const float* __restrict__ in, const float* __restrict__ g,
    uint32_t* __restrict__ out, int S, int Hn)
{
    int row  = blockIdx.x * 8 + (threadIdx.x >> 5);
    int lane = threadIdx.x & 31;
    int b    = row / (S * Hn);
    int rem  = row % (S * Hn);
    int s    = rem / Hn;
    int h    = rem % Hn;

    const float2 xv = *(const float2*)(in + (size_t)row * 64 + lane * 2);
    float ss = xv.x * xv.x + xv.y * xv.y;
    #pragma unroll
    for (int m = 1; m < 32; m <<= 1) ss += __shfl_xor(ss, m, 32);
    float r = rsqrtf(ss * (1.0f / 64.0f) + 1.1920929e-07f);
    float x0 = xv.x * r * g[lane * 2];
    float x1 = xv.y * r * g[lane * 2 + 1];
    out[((size_t)(b * Hn + h) * S + s) * 32 + lane] = pack2(x0, x1);
}

// ---------------------------------------------------------------------------
// Gene fitness per (b,kv): inv=1/(masked mean sigmoid(v)+0.5); normalize.
// ---------------------------------------------------------------------------
__global__ __launch_bounds__(256) void fitness_kernel(
    const float* __restrict__ vraw, const unsigned char* __restrict__ mask,
    float* __restrict__ fit, int S, int HKVc)
{
    const int b  = blockIdx.x / HKVc;
    const int kv = blockIdx.x % HKVc;
    const int t  = threadIdx.x;
    const int d  = t & 63, grp = t >> 6;

    __shared__ float red[256];
    __shared__ float invsh[64];

    float ms = 0.f;
    for (int s = t; s < S; s += 256) ms += mask[b * S + s] ? 1.f : 0.f;
    red[t] = ms;
    __syncthreads();
    for (int off = 128; off > 0; off >>= 1) {
        if (t < off) red[t] += red[t + off];
        __syncthreads();
    }
    const float msumT = red[0];
    __syncthreads();

    float acc = 0.f;
    for (int s = grp; s < S; s += 4) {
        float mv = mask[b * S + s] ? 1.f : 0.f;
        float x  = vraw[(((size_t)b * S + s) * HKVc + kv) * 64 + d];
        acc += mv / (1.f + exp2f(-x * 1.4426950408889634f));
    }
    red[t] = acc;
    __syncthreads();
    float inv = 0.f;
    if (t < 64) {
        float sum  = red[t] + red[t + 64] + red[t + 128] + red[t + 192];
        float mean = sum / fmaxf(msumT, 1.0f);
        inv = 1.0f / (mean + 0.5f);
        invsh[t] = inv;
    }
    __syncthreads();
    if (t < 64) {
        float tot = 0.f;
        #pragma unroll
        for (int i = 0; i < 64; ++i) tot += invsh[i];
        fit[(b * HKVc + kv) * 64 + t] = inv / tot;
    }
}

// ---------------------------------------------------------------------------
// Flash attention, double-buffered async K/V tiles (32 keys each).
// Qh (B,H,S,64) f16; Kh (B,HKV,S,64) f16; Vt (B,HKV,64,S) f16;
// O  (B,S,H*64) f16 (feeds output projection).
// ---------------------------------------------------------------------------
__global__ __launch_bounds__(256) void attn_kernel(
    const uint32_t* __restrict__ Qh, const uint16_t* __restrict__ Kh,
    const uint16_t* __restrict__ Vt, const float* __restrict__ fit,
    const unsigned char* __restrict__ mask, uint16_t* __restrict__ O,
    int S, int Hq, int HKVc)
{
    const int h  = blockIdx.y;
    const int b  = blockIdx.z;
    const int kv = h / (Hq / HKVc);
    const int t = threadIdx.x, lane = t & 31, wave = t >> 5;
    const int qbase = blockIdx.x * 128 + wave * 16;

    __shared__ __align__(16) uint32_t Kbuf[2][32 * 32];   // [key][32 dw of d]
    __shared__ __align__(16) uint32_t Vbuf[2][64 * 16];   // [d][16 dw of key]
    __shared__ __align__(16) uint32_t Plds[8 * 16 * 16];  // per-wave 16x32 halves

    const uint32_t* qp = Qh + (size_t)(b * Hq + h) * S * 32;
    const uint16_t* kp = Kh + (size_t)(b * HKVc + kv) * S * 64;
    const uint16_t* vp = Vt + (size_t)(b * HKVc + kv) * 64 * S;

    const uint32_t kBase = (uint32_t)(uintptr_t)&Kbuf[0][0];
    const uint32_t vBase = (uint32_t)(uintptr_t)&Vbuf[0][0];
    const uint64_t gK = (uint64_t)(uintptr_t)kp + (uint64_t)(t >> 3) * 128 + (t & 7) * 16;
    const uint64_t gV = (uint64_t)(uintptr_t)vp + (uint64_t)(t >> 2) * (size_t)S * 2 + (t & 3) * 16;
    const uint32_t ldsK = kBase + (uint32_t)t * 16;
    const uint32_t ldsV = vBase + (uint32_t)t * 16;

    // Q fragments (rows qbase..+15, D=64 split into two K=32 steps).
    v16h aq[2];
    {
        const uint32_t* rowp = qp + (size_t)(qbase + (lane & 15)) * 32;
        aq[0] = load_afrag(rowp, lane);
        aq[1] = load_afrag(rowp + 16, lane);
    }

    v8f acc[4];
    #pragma unroll
    for (int nb = 0; nb < 4; ++nb) acc[nb] = (v8f){0.f,0.f,0.f,0.f,0.f,0.f,0.f,0.f};
    float mrow[8], lrow[8];
    #pragma unroll
    for (int r = 0; r < 8; ++r) { mrow[r] = -1e30f; lrow[r] = 0.f; }

    const float sm = 0.125f * 1.44269504f;

    auto issue = [&](int buf, int kt) {
        const uint32_t boK = (uint32_t)buf * (32 * 32 * 4);
        const uint32_t boV = (uint32_t)buf * (64 * 16 * 4);
        async_b128(ldsK + boK, gK + (uint64_t)kt * 128);
        async_b128(ldsV + boV, gV + (uint64_t)kt * 2);
    };
    auto process = [&](int cur, int kt) {
        // Scores over 32 keys: QK^T (4 WMMAs).
        v8f s0 = (v8f){0.f,0.f,0.f,0.f,0.f,0.f,0.f,0.f};
        v8f s1 = s0;
        #pragma unroll
        for (int ds = 0; ds < 2; ++ds) {
            v16h b0 = load_bfrag(&Kbuf[cur][(      (lane & 15)) * 32 + ds * 16], lane);
            v16h b1 = load_bfrag(&Kbuf[cur][(16 +  (lane & 15)) * 32 + ds * 16], lane);
            s0 = wmma_f16(aq[ds], b0, s0);
            s1 = wmma_f16(aq[ds], b1, s1);
        }

        float mk0 = mask[b * S + kt      + (lane & 15)] ? 0.f : -1e30f;
        float mk1 = mask[b * S + kt + 16 + (lane & 15)] ? 0.f : -1e30f;

        _Float16* pws = (_Float16*)&Plds[wave * 256];
        #pragma unroll
        for (int r = 0; r < 8; ++r) {
            float x0 = s0[r] * sm + mk0;
            float x1 = s1[r] * sm + mk1;
            float mx = fmaxf(x0, x1);
            #pragma unroll
            for (int m = 1; m < 16; m <<= 1) mx = fmaxf(mx, __shfl_xor(mx, m, 32));
            float nm = fmaxf(mrow[r], mx);
            float corr = exp2f(mrow[r] - nm);
            float e0 = exp2f(x0 - nm);
            float e1 = exp2f(x1 - nm);
            float rs = e0 + e1;
            #pragma unroll
            for (int m = 1; m < 16; m <<= 1) rs += __shfl_xor(rs, m, 32);
            lrow[r] = lrow[r] * corr + rs;
            mrow[r] = nm;
            #pragma unroll
            for (int nb = 0; nb < 4; ++nb) acc[nb][r] *= corr;
            int m_loc = r + ((lane >> 4) << 3);
            pws[m_loc * 32 +      (lane & 15)] = (_Float16)e0;
            pws[m_loc * 32 + 16 + (lane & 15)] = (_Float16)e1;
        }

        // P as A-fragment; O += P @ V (4 WMMAs).
        v16h ap = load_afrag(&Plds[wave * 256 + (lane & 15) * 16], lane);
        #pragma unroll
        for (int nb = 0; nb < 4; ++nb) {
            v16h bv = load_bfrag(&Vbuf[cur][(nb * 16 + (lane & 15)) * 16], lane);
            acc[nb] = wmma_f16(ap, bv, acc[nb]);
        }
    };

    issue(0, 0);
    for (int kt = 0; kt < S - 32; kt += 32) {
        const int cur = (kt >> 5) & 1;
        issue(cur ^ 1, kt + 32);
        wait_async2();
        __syncthreads();
        process(cur, kt);
        __syncthreads();
    }
    {
        const int cur = ((S - 32) >> 5) & 1;
        wait_async0();
        __syncthreads();
        process(cur, S - 32);
    }

    // Epilogue: /l_i, * fitness, write f16 (B,S,H*D).
    #pragma unroll
    for (int nb = 0; nb < 4; ++nb) {
        int d = nb * 16 + (lane & 15);
        float fv = fit[(b * HKVc + kv) * 64 + d];
        #pragma unroll
        for (int r = 0; r < 8; ++r) {
            int m = qbase + r + ((lane >> 4) << 3);
            float val = (acc[nb][r] / lrow[r]) * fv;
            O[((size_t)(b * S + m) * Hq + h) * 64 + d] = f16bits(val);
        }
    }
}

// ---------------------------------------------------------------------------
// Host-side orchestration.
// ---------------------------------------------------------------------------
extern "C" void kernel_launch(void* const* d_in, const int* in_sizes, int n_in,
                              void* d_out, int out_size, void* d_ws, size_t ws_size,
                              hipStream_t stream) {
    (void)in_sizes; (void)n_in; (void)out_size; (void)ws_size;

    const float* x           = (const float*)d_in[0];
    const unsigned char* msk = (const unsigned char*)d_in[1];
    const float* Wq          = (const float*)d_in[2];
    const float* bq          = (const float*)d_in[3];
    const float* Wk          = (const float*)d_in[4];
    const float* bk          = (const float*)d_in[5];
    const float* Wv          = (const float*)d_in[6];
    const float* bv          = (const float*)d_in[7];
    const float* Wo          = (const float*)d_in[8];
    const float* bo          = (const float*)d_in[9];
    const float* gq          = (const float*)d_in[10];
    const float* gk          = (const float*)d_in[11];

    const int B = 2, S = 2048, E = 1024, H = 16, HKV = 4, D = 64;
    const int M = B * S;

    char* ws = (char*)d_ws;
    uint16_t* xh   = (uint16_t*)ws; ws += (size_t)M * E * 2;              //  8 MB
    uint16_t* Wqt  = (uint16_t*)ws; ws += (size_t)E * H * D * 2;          //  2 MB
    uint16_t* Wkt  = (uint16_t*)ws; ws += (size_t)E * HKV * D * 2;        // .5 MB
    uint16_t* Wvt  = (uint16_t*)ws; ws += (size_t)E * HKV * D * 2;        // .5 MB
    uint16_t* Wot  = (uint16_t*)ws; ws += (size_t)H * D * E * 2;          //  2 MB
    float*    qraw = (float*)ws;    ws += (size_t)M * H * D * 4;          // 16 MB
    float*    kraw = (float*)ws;    ws += (size_t)M * HKV * D * 4;        //  4 MB
    float*    vraw = (float*)ws;    ws += (size_t)M * HKV * D * 4;        //  4 MB
    uint32_t* qh   = (uint32_t*)ws; ws += (size_t)M * H * D * 2;          //  8 MB
    uint32_t* kh   = (uint32_t*)ws; ws += (size_t)M * HKV * D * 2;        //  2 MB
    uint16_t* vt   = (uint16_t*)ws; ws += (size_t)M * HKV * D * 2;        //  2 MB
    float*    fit  = (float*)ws;    ws += (size_t)B * HKV * D * 4;
    uint16_t* attno = (uint16_t*)qraw; // q raw consumed before attention

    dim3 blk(256);

    // f16 conversions (weights transposed so GEMM B-tiles are pure copies).
    f32_to_f16_kernel<<<(M * E / 2 + 255) / 256, blk, 0, stream>>>(x, (uint32_t*)xh, M * E / 2);
    transpose_f16_kernel<<<(E * H * D   + 255) / 256, blk, 0, stream>>>(Wq, Wqt, E, H * D);
    transpose_f16_kernel<<<(E * HKV * D + 255) / 256, blk, 0, stream>>>(Wk, Wkt, E, HKV * D);
    transpose_f16_kernel<<<(E * HKV * D + 255) / 256, blk, 0, stream>>>(Wv, Wvt, E, HKV * D);
    transpose_f16_kernel<<<(H * D * E   + 255) / 256, blk, 0, stream>>>(Wo, Wot, H * D, E);

    // QKV projections (block tile 128x128).
    gemm_wmma_kernel<<<dim3(M / 128, (H * D) / 128),   blk, 0, stream>>>(xh, Wqt, bq, qraw, M, H * D,   E);
    gemm_wmma_kernel<<<dim3(M / 128, (HKV * D) / 128), blk, 0, stream>>>(xh, Wkt, bk, kraw, M, HKV * D, E);
    gemm_wmma_kernel<<<dim3(M / 128, (HKV * D) / 128), blk, 0, stream>>>(xh, Wvt, bv, vraw, M, HKV * D, E);

    // Q/K RMSNorm to (B,h,S,D) f16; V to (B,kv,D,S) f16.
    rmsnorm_pack_kernel<<<(B * S * H)   / 8, blk, 0, stream>>>(qraw, gq, qh, S, H);
    rmsnorm_pack_kernel<<<(B * S * HKV) / 8, blk, 0, stream>>>(kraw, gk, kh, S, HKV);
    vpack_t_kernel<<<(M * HKV * D) / 256, blk, 0, stream>>>(vraw, vt, S, HKV);

    // Gene fitness.
    fitness_kernel<<<B * HKV, blk, 0, stream>>>(vraw, msk, fit, S, HKV);

    // Flash attention.
    attn_kernel<<<dim3(S / 128, H, B), blk, 0, stream>>>(qh, (const uint16_t*)kh, vt, fit, msk, attno, S, H, HKV);

    // Output projection.
    gemm_wmma_kernel<<<dim3(M / 128, E / 128), blk, 0, stream>>>(attno, Wot, bo, (float*)d_out, M, E, H * D);
}
